// mgfn_27453430956374
// MI455X (gfx1250) — matrix operand order repaired and checked by reading, not compiled, not run on hardware
//
#include <hip/hip_runtime.h>
#include <hip/hip_bf16.h>

typedef __attribute__((ext_vector_type(16))) _Float16 v16h;
typedef __attribute__((ext_vector_type(8)))  float    v8f;

#define T128 128
#define BTOT 320   // 32 * 10
#define EPSF 1e-5f

// ---------------------------------------------------------------------------
// WMMA tile helpers (v_wmma_f32_16x16x32_f16, wave32)
// A (16x32 f16): lane row = lane&15; K-half base = (lane&16)?16:0; element i ->
//   K = base + i
// B (32x16 f16): lane col = lane&15; K = ((lane&16)?16:0) + i
// C/D (16x16 f32): lane col N = lane&15; VGPR r -> M = r + ((lane&16)?8:0)
// ---------------------------------------------------------------------------

__device__ __forceinline__ v16h load_a_rowmajor(const float* __restrict__ W,
                                                int ldw, int row, int c0, int lane) {
  const float* wr = W + (size_t)row * ldw + c0 + ((lane & 16) ? 16 : 0);
  v16h a = {};
#pragma unroll
  for (int i = 0; i < 16; ++i) {
    int k = i;
    a[i] = (_Float16)wr[k];
  }
  return a;
}

__device__ __forceinline__ v16h load_b_strided(const float* __restrict__ base,
                                               size_t stride) {
  v16h b = {};
#pragma unroll
  for (int i = 0; i < 16; ++i) b[i] = (_Float16)base[(size_t)i * stride];
  return b;
}

// ---------------------------------------------------------------------------
// Pointwise-conv GEMM, 16(M) x 32(N) per wave, B tile staged through LDS
// (shared by all 4 waves): Y[b,o,t] = sum_c W[o,c]*X[b,c,t] (+bias)(+R)
// grid (O/64, 4, nB), block 128 (4 waves)
// ---------------------------------------------------------------------------
template <bool BIAS, bool RES>
__global__ __launch_bounds__(128) void k_gemm1_t(
    const float* __restrict__ X, const float* __restrict__ W,
    const float* __restrict__ bias, const float* __restrict__ R,
    float* __restrict__ Y, int C, int O, int b0x, int b0y) {
  __shared__ _Float16 ldsb[32 * 32];          // [t][c] f16
  const int lane = threadIdx.x & 31;
  const int wv   = threadIdx.x >> 5;
  const int n    = lane & 15;
  const int hi   = (lane & 16) ? 1 : 0;
  const int otile = (blockIdx.x * 4 + wv) * 16;
  const int ttile = blockIdx.y * 32;
  const int bz    = blockIdx.z;
  const int lt = threadIdx.x & 31;            // t-row handled by this thread
  const int lc = (threadIdx.x >> 5) * 8;      // c-segment handled by this thread

  const float* Xb = X + (size_t)(b0x + bz) * C * T128;
  const int arow = otile + n;
  const int kb16 = hi ? 16 : 0;
  v8f acc0 = {}, acc1 = {};
  for (int c0 = 0; c0 < C; c0 += 32) {
    __syncthreads();
    {
      const float* gp = Xb + (size_t)(c0 + lc) * T128 + ttile + lt;
#pragma unroll
      for (int j = 0; j < 8; ++j)
        ldsb[lt * 32 + lc + j] = (_Float16)gp[(size_t)j * T128];
    }
    __syncthreads();
    v16h a = load_a_rowmajor(W, C, arow, c0, lane);
    v16h bm0 = {}, bm1 = {};
    const _Float16* l0 = &ldsb[n * 32 + kb16];
#pragma unroll
    for (int i = 0; i < 16; ++i) { bm0[i] = l0[i]; bm1[i] = l0[16 * 32 + i]; }
    acc0 = __builtin_amdgcn_wmma_f32_16x16x32_f16(false, a, false, bm0,
                                                  (short)0, acc0, false, false);
    acc1 = __builtin_amdgcn_wmma_f32_16x16x32_f16(false, a, false, bm1,
                                                  (short)0, acc1, false, false);
  }
  const int t0 = ttile + n;
  const size_t ob = (size_t)(b0y + bz) * O * T128 + t0;
#pragma unroll
  for (int r = 0; r < 8; ++r) {
    int o = otile + r + hi * 8;
    float v0 = acc0[r], v1 = acc1[r];
    if (BIAS) { float bo = bias[o]; v0 += bo; v1 += bo; }
    size_t yi = ob + (size_t)o * T128;
    if (RES) { v0 += R[yi]; v1 += R[yi + 16]; }
    Y[yi] = v0;
    Y[yi + 16] = v1;
  }
}

// ---------------------------------------------------------------------------
// 3-tap conv GEMM with residual: Y = X + conv3(X,W3) + bias ; W3 is [O,C,3]
// 16 x 32 per wave; t-halo (34 cols) staged once in LDS and served to all
// 3 taps x 2 sub-tiles x 4 waves. grid (C/64, 4, nB)
// ---------------------------------------------------------------------------
__global__ __launch_bounds__(128) void k_gemm3(
    const float* __restrict__ X, const float* __restrict__ W3,
    const float* __restrict__ bias, float* __restrict__ Y, int C) {
  __shared__ _Float16 ldsb[36 * 32];          // [t-halo][c] f16, halo = ttile-1
  const int lane = threadIdx.x & 31;
  const int wv   = threadIdx.x >> 5;
  const int n    = lane & 15;
  const int hi   = (lane & 16) ? 1 : 0;
  const int otile = (blockIdx.x * 4 + wv) * 16;
  const int ttile = blockIdx.y * 32;
  const int bz    = blockIdx.z;
  const int lt = threadIdx.x & 31;
  const int lc = (threadIdx.x >> 5) * 8;
  const int O = C;

  const float* Xb = X + (size_t)bz * C * T128;
  const int arow = otile + n;
  const int kb16 = hi ? 16 : 0;
  v8f acc0 = {}, acc1 = {};
  for (int c0 = 0; c0 < C; c0 += 32) {
    __syncthreads();
#pragma unroll 2
    for (int p = 0; p < 2; ++p) {
      int t = lt + p * 32;
      if (t < 36) {
        int gt = ttile + t - 1;
        bool ok = ((unsigned)gt < 128u) && (t < 34);
        const float* gp = Xb + (size_t)(c0 + lc) * T128 + gt;
#pragma unroll
        for (int j = 0; j < 8; ++j)
          ldsb[t * 32 + lc + j] = ok ? (_Float16)gp[(size_t)j * T128] : (_Float16)0.f;
      }
    }
    __syncthreads();
    for (int k = 0; k < 3; ++k) {
      const float* wr = W3 + ((size_t)arow * C + c0 + (hi ? 16 : 0)) * 3 + k;
      v16h a = {};
#pragma unroll
      for (int i = 0; i < 16; ++i) {
        int kk = i;
        a[i] = (_Float16)wr[3 * kk];
      }
      v16h bm0 = {}, bm1 = {};
      const _Float16* l0 = &ldsb[(n + k) * 32 + kb16];
#pragma unroll
      for (int i = 0; i < 16; ++i) { bm0[i] = l0[i]; bm1[i] = l0[16 * 32 + i]; }
      acc0 = __builtin_amdgcn_wmma_f32_16x16x32_f16(false, a, false, bm0,
                                                    (short)0, acc0, false, false);
      acc1 = __builtin_amdgcn_wmma_f32_16x16x32_f16(false, a, false, bm1,
                                                    (short)0, acc1, false, false);
    }
  }
  const int t0 = ttile + n;
  const size_t ob = (size_t)bz * O * T128 + t0;
#pragma unroll
  for (int r = 0; r < 8; ++r) {
    int o = otile + r + hi * 8;
    float bo = bias[o];
    size_t yi = ob + (size_t)o * T128;
    Y[yi]      = acc0[r] + bo + X[yi];
    Y[yi + 16] = acc1[r] + bo + X[yi + 16];
  }
}

// ---------------------------------------------------------------------------
// Input conv: Y[b,o,t] = conv3_{2048ch}(video) + bias + 0.1*magconv
// video layout: video[(b*128 + t)*2049 + c], mag channel c=2048.
// grid (1, 4, 320), block 128 (O=64 -> 4 o-tiles = 4 waves; 32 t per wave)
// ---------------------------------------------------------------------------
__global__ __launch_bounds__(128) void k_inconv(
    const float* __restrict__ video, const float* __restrict__ tw,
    const float* __restrict__ tb, const float* __restrict__ mw,
    const float* __restrict__ mb, float* __restrict__ Y) {
  const int lane = threadIdx.x & 31;
  const int wv   = threadIdx.x >> 5;
  const int n    = lane & 15;
  const int hi   = (lane & 16) ? 1 : 0;
  const int otile = wv * 16;
  const int ttile = blockIdx.y * 32;
  const int bz    = blockIdx.z;

  const int arow = otile + n;
  v8f acc0 = {}, acc1 = {};
  for (int c0 = 0; c0 < 2048; c0 += 32) {
    for (int k = 0; k < 3; ++k) {
      const float* wr = tw + ((size_t)arow * 2048 + c0 + (hi ? 16 : 0)) * 3 + k;
      v16h a = {};
#pragma unroll
      for (int i = 0; i < 16; ++i) {
        int kk = i;
        a[i] = (_Float16)wr[3 * kk];
      }
      int tc0 = ttile + n + k - 1;
      int tc1 = tc0 + 16;
      v16h bm0 = {}, bm1 = {};
      if ((unsigned)tc0 < 128u) {
        const float* bp = video + ((size_t)bz * T128 + tc0) * 2049 + c0 + (hi ? 16 : 0);
#pragma unroll
        for (int i = 0; i < 16; ++i) bm0[i] = (_Float16)bp[i];
      }
      if ((unsigned)tc1 < 128u) {
        const float* bp = video + ((size_t)bz * T128 + tc1) * 2049 + c0 + (hi ? 16 : 0);
#pragma unroll
        for (int i = 0; i < 16; ++i) bm1[i] = (_Float16)bp[i];
      }
      acc0 = __builtin_amdgcn_wmma_f32_16x16x32_f16(false, a, false, bm0,
                                                    (short)0, acc0, false, false);
      acc1 = __builtin_amdgcn_wmma_f32_16x16x32_f16(false, a, false, bm1,
                                                    (short)0, acc1, false, false);
    }
  }
#pragma unroll 2
  for (int p = 0; p < 2; ++p) {
    const int t = ttile + p * 16 + n;
#pragma unroll
    for (int r = 0; r < 8; ++r) {
      int o = otile + r + hi * 8;
      float mag = mb[o];
      for (int k = 0; k < 3; ++k) {
        int tc = t + k - 1;
        if ((unsigned)tc < 128u)
          mag += mw[o * 3 + k] * video[((size_t)bz * T128 + tc) * 2049 + 2048];
      }
      float v = (p == 0 ? acc0[r] : acc1[r]) + tb[o] + 0.1f * mag;
      Y[((size_t)bz * 64 + o) * T128 + t] = v;
    }
  }
}

// ---------------------------------------------------------------------------
// Glance attention (stage 1: heads=1, d=64, T=128). qkv: [b,192,128]
// ---------------------------------------------------------------------------
__global__ __launch_bounds__(128) void k_sim(const float* __restrict__ qkv,
                                             float* __restrict__ sim) {
  const int lane = threadIdx.x & 31;
  const int wv   = threadIdx.x >> 5;
  const int n    = lane & 15;
  const int hi   = (lane & 16) ? 1 : 0;
  const int itile = (blockIdx.x * 4 + wv) * 16;
  const int jtile = blockIdx.y * 16;
  const int b     = blockIdx.z;
  const float* q = qkv + (size_t)b * 192 * T128;
  v8f acc = {};
  for (int d0 = 0; d0 < 64; d0 += 32) {
    v16h a = {};
#pragma unroll
    for (int i = 0; i < 16; ++i) {
      int k = (hi ? 16 : 0) + i;
      a[i] = (_Float16)(0.125f * q[(size_t)(d0 + k) * T128 + itile + n]);
    }
    const float* kp = q + (size_t)(64 + d0 + (hi ? 16 : 0)) * T128 + jtile + n;
    v16h bm = load_b_strided(kp, T128);
    acc = __builtin_amdgcn_wmma_f32_16x16x32_f16(false, a, false, bm,
                                                 (short)0, acc, false, false);
  }
#pragma unroll
  for (int r = 0; r < 8; ++r) {
    int i = itile + r + hi * 8;
    sim[((size_t)b * T128 + i) * T128 + jtile + n] = acc[r];
  }
}

__global__ void k_softmax(float* __restrict__ sim) {
  int id = blockIdx.x * blockDim.x + threadIdx.x;
  if (id >= BTOT * T128) return;
  float* row = sim + (size_t)id * T128;
  float mx = row[0];
  for (int j = 1; j < T128; ++j) mx = fmaxf(mx, row[j]);
  float s = 0.f;
  for (int j = 0; j < T128; ++j) s += expf(row[j] - mx);
  float inv = 1.f / s;
  for (int j = 0; j < T128; ++j) row[j] = expf(row[j] - mx) * inv;
}

// out[b,d,i] = sum_j attn[b,i,j] * v[d,j] ; v contiguous over j, attn over j.
__global__ __launch_bounds__(128) void k_av(const float* __restrict__ qkv,
                                            const float* __restrict__ attn,
                                            float* __restrict__ out) {
  const int lane = threadIdx.x & 31;
  const int wv   = threadIdx.x >> 5;
  const int n    = lane & 15;
  const int hi   = (lane & 16) ? 1 : 0;
  const int dtile = wv * 16;
  const int itile = blockIdx.y * 16;
  const int b     = blockIdx.z;
  const float* v = qkv + (size_t)b * 192 * T128 + 128 * T128;
  v8f acc = {};
  for (int j0 = 0; j0 < 128; j0 += 32) {
    const float* ar = v + (size_t)(dtile + n) * T128 + j0 + (hi ? 16 : 0);
    v16h a = {};
#pragma unroll
    for (int i = 0; i < 16; ++i) a[i] = (_Float16)ar[i];
    const float* br = attn + ((size_t)b * T128 + itile + n) * T128 + j0 + (hi ? 16 : 0);
    v16h bm = {};
#pragma unroll
    for (int i = 0; i < 16; ++i) bm[i] = (_Float16)br[i];
    acc = __builtin_amdgcn_wmma_f32_16x16x32_f16(false, a, false, bm,
                                                 (short)0, acc, false, false);
  }
#pragma unroll
  for (int r = 0; r < 8; ++r) {
    int d = dtile + r + hi * 8;
    out[((size_t)b * 64 + d) * T128 + itile + n] = acc[r];
  }
}

// ---------------------------------------------------------------------------
// Normalizations / elementwise
// ---------------------------------------------------------------------------
__global__ void k_lnch(const float* __restrict__ X, const float* __restrict__ g,
                       const float* __restrict__ bb, float* __restrict__ Y, int C) {
  int id = blockIdx.x * blockDim.x + threadIdx.x;
  if (id >= BTOT * T128) return;
  int b = id >> 7, t = id & 127;
  const float* xp = X + (size_t)b * C * T128 + t;
  float s = 0.f, s2 = 0.f;
  for (int c = 0; c < C; ++c) { float v = xp[(size_t)c * T128]; s += v; s2 += v * v; }
  float mean = s / C;
  float inv  = rsqrtf(s2 / C - mean * mean + EPSF);
  float* yp = Y + (size_t)b * C * T128 + t;
  for (int c = 0; c < C; ++c)
    yp[(size_t)c * T128] = (xp[(size_t)c * T128] - mean) * inv * g[c] + bb[c];
}

__global__ void k_bn(const float* __restrict__ X, const float* __restrict__ g,
                     const float* __restrict__ bb, const float* __restrict__ m,
                     const float* __restrict__ v, float* __restrict__ Y,
                     int C, size_t total) {
  size_t id = (size_t)blockIdx.x * blockDim.x + threadIdx.x;
  if (id >= total) return;
  int c = (int)((id >> 7) % C);
  Y[id] = (X[id] - m[c]) * rsqrtf(v[c] + EPSF) * g[c] + bb[c];
}

__global__ void k_gelu(float* __restrict__ X, size_t total) {
  size_t id = (size_t)blockIdx.x * blockDim.x + threadIdx.x;
  if (id >= total) return;
  float x = X[id];
  X[id] = 0.5f * x * (1.f + erff(x * 0.70710678118654752f));
}

// depthwise 5-tap over t, channel ch -> head hh = ch % heads
__global__ void k_rel(const float* __restrict__ V, const float* __restrict__ rw,
                      const float* __restrict__ rb, float* __restrict__ Y,
                      int inner, int heads, size_t total) {
  size_t id = (size_t)blockIdx.x * blockDim.x + threadIdx.x;
  if (id >= total) return;
  int t  = (int)(id & 127);
  int ch = (int)((id >> 7) % inner);
  int hh = ch % heads;
  float acc = rb[hh];
#pragma unroll
  for (int k = 0; k < 5; ++k) {
    int tc = t + k - 2;
    if ((unsigned)tc < 128u) acc += rw[hh * 5 + k] * V[id + (tc - t)];
  }
  Y[id] = acc;
}

// ---------------------------------------------------------------------------
// Head: feature LN (over f=1024) + fc-dot + L2 norm, crop means, top-2, gather
// ---------------------------------------------------------------------------
__global__ void k_feats(const float* __restrict__ X, const float* __restrict__ lg,
                        const float* __restrict__ lb, const float* __restrict__ fcw,
                        const float* __restrict__ fcb, float* __restrict__ feats,
                        float* __restrict__ scores, float* __restrict__ fnorm) {
  int id = blockIdx.x * blockDim.x + threadIdx.x;
  if (id >= BTOT * T128) return;
  int b = id >> 7, t = id & 127;
  const float* xp = X + (size_t)b * 1024 * T128 + t;
  float s = 0.f, s2 = 0.f;
  for (int f = 0; f < 1024; ++f) { float v = xp[(size_t)f * T128]; s += v; s2 += v * v; }
  float mean = s / 1024.f;
  float inv  = rsqrtf(s2 / 1024.f - mean * mean + EPSF);
  float* fr = feats + (size_t)id * 1024;
  float dot = 0.f, ss = 0.f;
  for (int f = 0; f < 1024; ++f) {
    float v = (xp[(size_t)f * T128] - mean) * inv * lg[f] + lb[f];
    fr[f] = v;
    dot += v * fcw[f];
    ss  += v * v;
  }
  scores[id] = 1.f / (1.f + expf(-(dot + fcb[0])));
  fnorm[id]  = sqrtf(ss);
}

__global__ void k_crop(const float* __restrict__ scores, const float* __restrict__ fnorm,
                       float* __restrict__ s_out, float* __restrict__ fmag) {
  int id = blockIdx.x * blockDim.x + threadIdx.x;
  if (id >= 32 * T128) return;
  int bi = id >> 7, t = id & 127;
  float ss = 0.f, mm = 0.f;
  for (int c = 0; c < 10; ++c) {
    int b = bi * 10 + c;
    ss += scores[b * T128 + t];
    mm += fnorm[b * T128 + t];
  }
  s_out[id] = ss * 0.1f;
  fmag[id]  = mm * 0.1f;
}

__global__ void k_topk(const float* __restrict__ fmag, const float* __restrict__ s,
                       float* __restrict__ score_ab, float* __restrict__ score_no,
                       int* __restrict__ idx) {
  int r = threadIdx.x;
  if (r >= 32) return;
  const float* row = fmag + r * T128;
  int i1 = 0; float v1 = row[0];
  for (int t = 1; t < T128; ++t) if (row[t] > v1) { v1 = row[t]; i1 = t; }
  int i2 = (i1 == 0) ? 1 : 0; float v2 = row[i2];
  for (int t = 0; t < T128; ++t)
    if (t != i1 && row[t] > v2) { v2 = row[t]; i2 = t; }
  float sc = 0.5f * (s[r * T128 + i1] + s[r * T128 + i2]);
  idx[r * 2] = i1; idx[r * 2 + 1] = i2;
  if (r < 16) score_no[r] = sc; else score_ab[r - 16] = sc;
}

__global__ void k_gather(const float* __restrict__ feats, const int* __restrict__ idx,
                         float* __restrict__ abn, float* __restrict__ nor) {
  int gid = blockIdx.x * blockDim.x + threadIdx.x;
  if (gid >= 2 * 327680) return;
  int half = gid / 327680;      // 0 = abnormal, 1 = normal
  int rem  = gid % 327680;
  int f    = rem % 1024;
  int k    = (rem / 1024) % 2;
  int rowi = rem / 2048;        // crop*16 + i
  int crop = rowi / 16, i = rowi % 16;
  int srcR = (half == 0) ? (16 + i) : i;
  int t    = idx[srcR * 2 + k];
  int b    = ((half == 0) ? 160 : 0) + i * 10 + crop;
  float v  = feats[((size_t)b * T128 + t) * 1024 + f];
  (half == 0 ? abn : nor)[rem] = v;
}

// ---------------------------------------------------------------------------
// Host side
// ---------------------------------------------------------------------------
struct GbAttn { const float *ln_b, *ln_g, *out_b, *out_w, *qkv_w; };
struct FbAttn { const float *bn_b, *bn_g, *bn_m, *bn_v, *out_b, *out_w, *rel_b, *rel_w, *v_w; };
struct FFp    { const float *b1, *b2, *ln_b, *ln_g, *w1, *w2; };
struct LayerP { GbAttn gb; FbAttn fb; FFp ff; const float *scc_b, *scc_w; };
struct DownP  { const float *b, *ln_b, *ln_g, *w; bool present; };

extern "C" void kernel_launch(void* const* d_in, const int* in_sizes, int n_in,
                              void* d_out, int out_size, void* d_ws, size_t ws_size,
                              hipStream_t stream) {
  (void)in_sizes; (void)n_in; (void)out_size; (void)ws_size;
  const int dims[3]   = {64, 128, 1024};
  const int depths[3] = {3, 3, 2};

  // ---- unpack params (jax pytree: sorted dict keys, None dropped) ----
  int pi = 1;
  auto P = [&]() { return (const float*)d_in[pi++]; };
  const float* video  = (const float*)d_in[0];
  const float* fc_b   = P();
  const float* fc_w   = P();
  const float* g_ln_b = P();
  const float* g_ln_g = P();
  DownP downs[3]; LayerP lay[3][3];
  for (int s = 0; s < 3; ++s) {
    if (s < 2) { downs[s].present = true; downs[s].b = P(); downs[s].ln_b = P(); downs[s].ln_g = P(); downs[s].w = P(); }
    else downs[s].present = false;
    for (int l = 0; l < depths[s]; ++l) {
      LayerP& L = lay[s][l];
      if (s == 0) { L.gb.ln_b = P(); L.gb.ln_g = P(); L.gb.out_b = P(); L.gb.out_w = P(); L.gb.qkv_w = P(); }
      else { L.fb.bn_b = P(); L.fb.bn_g = P(); L.fb.bn_m = P(); L.fb.bn_v = P();
             L.fb.out_b = P(); L.fb.out_w = P(); L.fb.rel_b = P(); L.fb.rel_w = P(); L.fb.v_w = P(); }
      L.ff.b1 = P(); L.ff.b2 = P(); L.ff.ln_b = P(); L.ff.ln_g = P(); L.ff.w1 = P(); L.ff.w2 = P();
      L.scc_b = P(); L.scc_w = P();
    }
  }
  const float* mag_b = P();
  const float* mag_w = P();
  const float* tok_b = P();
  const float* tok_w = P();

  // ---- workspace carve ----
  const size_t S1 = (size_t)BTOT * 1024 * T128;   // 41.94M floats
  float* bufA = (float*)d_ws;
  float* bufC = bufA + S1;
  float* bufB = bufC + S1;
  float* bufH = bufB + S1;                        // also 80*4096*128 == S1
  float* simb = bufH + S1;                        // 320*128*128
  float* scoresb = simb + (size_t)BTOT * T128 * T128;
  float* fnormb  = scoresb + BTOT * T128;
  float* fmagb   = fnormb + BTOT * T128;
  int*   idxb    = (int*)(fmagb + 32 * T128);

  float* cur = bufA;
  float* alt = bufC;

  const dim3 blk(128);
  const int LN_G = (BTOT * T128 + 255) / 256;

  // ---- input conv + magnitude branch ----
  k_inconv<<<dim3(1, 4, BTOT), blk, 0, stream>>>(video, tok_w, tok_b, mag_w, mag_b, cur);

  // ---- stages ----
  for (int s = 0; s < 3; ++s) {
    int dim = dims[s], hid = dim * 4, heads = dim / 64;
    size_t tot = (size_t)BTOT * dim * T128;
    for (int l = 0; l < depths[s]; ++l) {
      LayerP& L = lay[s][l];
      // scc: x = x + conv3(x)
      k_gemm3<<<dim3(dim / 64, 4, BTOT), blk, 0, stream>>>(cur, L.scc_w, L.scc_b, alt, dim);
      { float* t_ = cur; cur = alt; alt = t_; }
      // attention
      if (s == 0) {
        k_lnch<<<LN_G, 256, 0, stream>>>(cur, L.gb.ln_g, L.gb.ln_b, bufB, dim);
        k_gemm1_t<false, false><<<dim3(192 / 64, 4, BTOT), blk, 0, stream>>>(
            bufB, L.gb.qkv_w, nullptr, nullptr, bufH, dim, 192, 0, 0);
        k_sim<<<dim3(2, 8, BTOT), blk, 0, stream>>>(bufH, simb);
        k_softmax<<<LN_G, 256, 0, stream>>>(simb);
        k_av<<<dim3(1, 8, BTOT), blk, 0, stream>>>(bufH, simb, bufB);
        k_gemm1_t<true, true><<<dim3(dim / 64, 4, BTOT), blk, 0, stream>>>(
            bufB, L.gb.out_w, L.gb.out_b, cur, alt, 64, dim, 0, 0);
      } else {
        k_bn<<<(int)((tot + 255) / 256), 256, 0, stream>>>(cur, L.fb.bn_g, L.fb.bn_b, L.fb.bn_m, L.fb.bn_v, bufB, dim, tot);
        k_gemm1_t<false, false><<<dim3(dim / 64, 4, BTOT), blk, 0, stream>>>(
            bufB, L.fb.v_w, nullptr, nullptr, bufH, dim, dim, 0, 0);
        k_rel<<<(int)((tot + 255) / 256), 256, 0, stream>>>(bufH, L.fb.rel_w, L.fb.rel_b, bufB, dim, heads, tot);
        k_gemm1_t<true, true><<<dim3(dim / 64, 4, BTOT), blk, 0, stream>>>(
            bufB, L.fb.out_w, L.fb.out_b, cur, alt, dim, dim, 0, 0);
      }
      { float* t_ = cur; cur = alt; alt = t_; }
      // feedforward (batch-chunked so the 4096-ch hidden fits in bufH)
      k_lnch<<<LN_G, 256, 0, stream>>>(cur, L.ff.ln_g, L.ff.ln_b, bufB, dim);
      for (int c = 0; c < 4; ++c) {
        int b0 = c * 80;
        size_t htot = (size_t)80 * hid * T128;
        k_gemm1_t<true, false><<<dim3(hid / 64, 4, 80), blk, 0, stream>>>(
            bufB, L.ff.w1, L.ff.b1, nullptr, bufH, dim, hid, b0, 0);
        k_gelu<<<(int)((htot + 255) / 256), 256, 0, stream>>>(bufH, htot);
        k_gemm1_t<true, true><<<dim3(dim / 64, 4, 80), blk, 0, stream>>>(
            bufH, L.ff.w2, L.ff.b2, cur, alt, hid, dim, 0, b0);
      }
      { float* t_ = cur; cur = alt; alt = t_; }
    }
    if (downs[s].present) {
      int nd = dims[s + 1];
      k_lnch<<<LN_G, 256, 0, stream>>>(cur, downs[s].ln_g, downs[s].ln_b, bufB, dim);
      k_gemm1_t<true, false><<<dim3(nd / 64, 4, BTOT), blk, 0, stream>>>(
          bufB, downs[s].w, downs[s].b, nullptr, alt, dim, nd, 0, 0);
      { float* t_ = cur; cur = alt; alt = t_; }
    }
  }

  // ---- head ----
  float* feats = bufH;                    // 320*128*1024 == S1, bufH free now
  float* out   = (float*)d_out;
  float* out_ab = out;                    // (16,1)
  float* out_no = out + 16;               // (16,1)
  float* out_abn = out + 32;              // (160,2,1024)
  float* out_nor = out + 32 + 327680;     // (160,2,1024)
  float* out_s   = out + 32 + 2 * 327680; // (32,128,1)

  k_feats<<<LN_G, 256, 0, stream>>>(cur, g_ln_g, g_ln_b, fc_w, fc_b, feats, scoresb, fnormb);
  k_crop<<<16, 256, 0, stream>>>(scoresb, fnormb, out_s, fmagb);
  k_topk<<<1, 32, 0, stream>>>(fmagb, out_s, out_ab, out_no, idxb);
  k_gather<<<2560, 256, 0, stream>>>(feats, idxb, out_abn, out_nor);
}
